// RelativeSelfAttention2d_71949292142681
// MI455X (gfx1250) — compile-verified, matrix operand order checked by reading, not run on hardware
//
#include <hip/hip_runtime.h>
#include <hip/hip_bf16.h>

// ---------------------------------------------------------------------------
// Types for CDNA5 WMMA (bf16 in, f32 out), wave32.
// ---------------------------------------------------------------------------
typedef __bf16 bf16;
typedef __attribute__((ext_vector_type(16))) __bf16 v16bf;
typedef __attribute__((ext_vector_type(8)))  __bf16 v8bf;
typedef __attribute__((ext_vector_type(2)))  __bf16 v2bf;
typedef __attribute__((ext_vector_type(8)))  float  v8f;

#define BATCH     8
#define CCH       384
#define NTOK      1024          // H*W = 32*32
#define NHEADS    8
#define HDIM      48
#define QKDIM     64            // HDIM padded to multiple of 32 for K-steps
#define VROWS     64            // V^T rows: 48 data + ones row (48) + zeros
#define HIDDEN    192
#define TBL       63            // relative offsets per axis: [-31,31]

__device__ __forceinline__ v8f wmma_bf16(v16bf a, v16bf b, v8f c) {
    return __builtin_amdgcn_wmma_f32_16x16x32_bf16(
        false, a, false, b, (short)0, c, false, false);
}

// Build one 16x32 A/B fragment per ISA layout: lane half selects K group
// {0..7,16..23} vs {8..15,24..31}; caller passes the two 16-byte chunks.
__device__ __forceinline__ v16bf load_frag(const bf16* p0, const bf16* p1) {
    v8bf lo = *(const v8bf*)p0;
    v8bf hi = *(const v8bf*)p1;
    v16bf r;
#pragma unroll
    for (int i = 0; i < 8; ++i) { r[i] = lo[i]; r[i + 8] = hi[i]; }
    return r;
}

// Max-reduce across each 16-lane group (C-layout row group) with DPP
// row_ror rotations: pure VALU, no LDS traffic, result in all 16 lanes.
__device__ __forceinline__ float red16_max(float v) {
    int x;
    x = __builtin_amdgcn_update_dpp(0, __float_as_int(v), 0x128, 0xf, 0xf, true); // row_ror:8
    v = fmaxf(v, __int_as_float(x));
    x = __builtin_amdgcn_update_dpp(0, __float_as_int(v), 0x124, 0xf, 0xf, true); // row_ror:4
    v = fmaxf(v, __int_as_float(x));
    x = __builtin_amdgcn_update_dpp(0, __float_as_int(v), 0x122, 0xf, 0xf, true); // row_ror:2
    v = fmaxf(v, __int_as_float(x));
    x = __builtin_amdgcn_update_dpp(0, __float_as_int(v), 0x121, 0xf, 0xf, true); // row_ror:1
    v = fmaxf(v, __int_as_float(x));
    return v;
}

// ---------------------------------------------------------------------------
// Small prep kernels
// ---------------------------------------------------------------------------
__global__ void convert_f32_bf16(const float* __restrict__ s,
                                 bf16* __restrict__ d, int n) {
    int i = blockIdx.x * blockDim.x + threadIdx.x;
    if (i < n) d[i] = (bf16)s[i];
}

// x[B,C,N] -> tokens bf16 [B,N,C]
__global__ void pack_tokens(const float* __restrict__ x, bf16* __restrict__ t) {
    int i = blockIdx.x * blockDim.x + threadIdx.x;
    if (i >= BATCH * NTOK * CCH) return;
    int c = i % CCH;
    int n = (i / CCH) % NTOK;
    int b = i / (CCH * NTOK);
    t[i] = (bf16)x[((long)b * CCH + c) * NTOK + n];
}

// Relative-position bias table: only 63*63 distinct (dh,dw) offsets exist.
// tbl[(dh+31)*63 + (dw+31)][head], exact erf-GELU as in the reference.
__global__ void bias_table_kernel(const float* __restrict__ w1,
                                  const float* __restrict__ b1,
                                  const float* __restrict__ w2,
                                  const float* __restrict__ b2,
                                  float* __restrict__ tbl) {
    int idx = blockIdx.x * blockDim.x + threadIdx.x;
    if (idx >= TBL * TBL) return;
    const float r0 = (float)(idx / TBL - 31) * (1.0f / 31.0f);
    const float r1 = (float)(idx % TBL - 31) * (1.0f / 31.0f);
    float acc[NHEADS] = {};
    for (int j = 0; j < HIDDEN; ++j) {
        float h = r0 * w1[j] + r1 * w1[HIDDEN + j] + b1[j];
        h = 0.5f * h * (1.0f + erff(h * 0.70710678118f));
#pragma unroll
        for (int t = 0; t < NHEADS; ++t) acc[t] += h * w2[j * NHEADS + t];
    }
#pragma unroll
    for (int t = 0; t < NHEADS; ++t) tbl[idx * NHEADS + t] = acc[t] + b2[t];
}

// qkv f32 [B,N,3,heads,48] -> Q*scale bf16 [B,h,N,64] (zero-padded),
// K bf16 [B,h,N,64] (zero-padded), V^T bf16 [B,h,64,N] where row 48 is all
// ones (so P@V also accumulates the softmax denominator) and 49..63 are 0.
__global__ void split_qkv_kernel(const float* __restrict__ qkv,
                                 bf16* __restrict__ Qo,
                                 bf16* __restrict__ Ko,
                                 bf16* __restrict__ Vt) {
    int idx = blockIdx.x * blockDim.x + threadIdx.x;   // 2^22 threads
    const int dd = idx & 63;
    const int n  = (idx >> 6) & 1023;
    const int h  = (idx >> 16) & 7;
    const int b  = idx >> 19;
    if (b >= BATCH) return;
    const float scale = 0.14433756729740646f;          // 48^-0.5
    const long base = ((long)(b * NTOK + n)) * (3 * CCH) + h * HDIM + dd;
    const long vslot = (((long)(b * NHEADS + h) * VROWS) + dd) * NTOK + n;
    if (dd < HDIM) {
        Qo[idx] = (bf16)(qkv[base] * scale);
        Ko[idx] = (bf16)qkv[base + CCH];
        Vt[vslot] = (bf16)qkv[base + 2 * CCH];
    } else {
        Qo[idx] = (bf16)0.0f;
        Ko[idx] = (bf16)0.0f;
        Vt[vslot] = (dd == HDIM) ? (bf16)1.0f : (bf16)0.0f;  // ones column
    }
}

// ---------------------------------------------------------------------------
// Tiled WMMA GEMM: C[M,N] = A[M,K]*B[K,N] + bias, bf16 inputs, f32 out.
// Block tile 64x128, BK=32, 8 waves in 2x4, each wave 2x2 WMMA tiles.
// Double-buffered LDS (register prefetch pipeline, 1 barrier per K-step).
// LDS pitch 40 bf16 (80B): 16B-aligned b128 fragment loads, conflict-free.
// B staged as k-pairs so transposed LDS writes are packed b32 (full width).
// N, K, TRANS are compile-time; TRANS stores out[col*ldT + row] with b128s.
// ---------------------------------------------------------------------------
#define GBM 64
#define GBN 128
#define GBK 32
#define LPITCH 40

template <int N, int K, bool TRANS>
__global__ __launch_bounds__(256, 2)
void gemm_bf16_kernel(const bf16* __restrict__ A, const bf16* __restrict__ Bw,
                      const float* __restrict__ bias, float* __restrict__ Cout,
                      long sA, long sC, int ldT) {
    __shared__ bf16 AsB[2][GBM * LPITCH];
    __shared__ bf16 BtB[2][GBN * LPITCH];

    const int tid  = threadIdx.x;
    const int lane = tid & 31;
    const int wv   = tid >> 5;
    constexpr int tilesN = N / GBN;
    const int row0 = (blockIdx.x / tilesN) * GBM;
    const int col0 = (blockIdx.x % tilesN) * GBN;
    A    += (long)blockIdx.y * sA;
    Cout += (long)blockIdx.y * sC;

    const int wm = wv >> 2, wn = wv & 3;
    const int halfId = lane >> 4, l16 = lane & 15, kb = halfId * 8;

    // staging maps
    const int ar = tid >> 2,  ac = (tid & 3) * 8;     // A: 64x32, 8 elem/thr
    const int k2 = (tid >> 4) * 2;                    // B: k-row pair 0..30
    const int nb = (tid & 15) * 8;                    // B: 8 n-columns

    v8f acc[2][2] = {};

    uint4 aReg;
    v8bf  bReg0, bReg1;   // rows k2 and k2+1, 8 n-columns each

    auto loadGlobal = [&](int k0) {
        aReg = *(const uint4*)(A + (long)(row0 + ar) * K + k0 + ac);
        const bf16* src = Bw + (long)(k0 + k2) * N + col0 + nb;
        bReg0 = *(const v8bf*)(src);
        bReg1 = *(const v8bf*)(src + N);
    };
    auto storeLDS = [&](int buf) {
        *(uint4*)(&AsB[buf][ar * LPITCH + ac]) = aReg;
#pragma unroll
        for (int i = 0; i < 8; ++i) {
            v2bf pk; pk[0] = bReg0[i]; pk[1] = bReg1[i];   // {k2, k2+1}
            *(v2bf*)(&BtB[buf][(nb + i) * LPITCH + k2]) = pk;
        }
    };
    auto compute = [&](int buf) {
        const bf16* As = AsB[buf];
        const bf16* Bt = BtB[buf];
        v16bf af[2], bfrag[2];
#pragma unroll
        for (int i = 0; i < 2; ++i) {
            const int r = wm * 32 + i * 16 + l16;
            af[i] = load_frag(&As[r * LPITCH + kb], &As[r * LPITCH + kb + 16]);
        }
#pragma unroll
        for (int j = 0; j < 2; ++j) {
            const int c = wn * 32 + j * 16 + l16;
            bfrag[j] = load_frag(&Bt[c * LPITCH + kb], &Bt[c * LPITCH + kb + 16]);
        }
#pragma unroll
        for (int i = 0; i < 2; ++i)
#pragma unroll
            for (int j = 0; j < 2; ++j)
                acc[i][j] = wmma_bf16(af[i], bfrag[j], acc[i][j]);
    };

    constexpr int KSTEPS = K / GBK;
    loadGlobal(0);
#pragma unroll 2
    for (int kt = 0; kt < KSTEPS - 1; ++kt) {
        storeLDS(kt & 1);
        __syncthreads();
        loadGlobal((kt + 1) * GBK);   // overlap next tile's loads with WMMAs
        compute(kt & 1);
    }
    storeLDS((KSTEPS - 1) & 1);
    __syncthreads();
    compute((KSTEPS - 1) & 1);

    // epilogue (branch-free: TRANS is compile-time, bias always present)
#pragma unroll
    for (int i = 0; i < 2; ++i)
#pragma unroll
        for (int j = 0; j < 2; ++j) {
            const int col = col0 + wn * 32 + j * 16 + l16;
            const int rowBase = row0 + wm * 32 + i * 16 + 8 * halfId;
            const float bc = bias[col];
            if constexpr (TRANS) {
                // 8 accumulator rows are contiguous along the transposed dim
                float4 lo, hi;
                lo.x = acc[i][j][0] + bc; lo.y = acc[i][j][1] + bc;
                lo.z = acc[i][j][2] + bc; lo.w = acc[i][j][3] + bc;
                hi.x = acc[i][j][4] + bc; hi.y = acc[i][j][5] + bc;
                hi.z = acc[i][j][6] + bc; hi.w = acc[i][j][7] + bc;
                float* p = Cout + (long)col * ldT + rowBase;
                *(float4*)(p)     = lo;
                *(float4*)(p + 4) = hi;
            } else {
                float* p = Cout + (long)rowBase * N + col;
#pragma unroll
                for (int r = 0; r < 8; ++r)
                    p[(long)r * N] = acc[i][j][r] + bc;
            }
        }
}

// ---------------------------------------------------------------------------
// Flash-style attention, one wave per 16 query rows, 32 keys per step.
// S = Q@K^T (2 chained K=32 WMMAs, d padded to 64) + table bias.
// Online softmax: row-max via DPP rotate reduction (pure VALU); the row-sum
// rides in a 4th P@V output tile against V's ones-column, so it is rescaled
// by corr together with O and needs no per-step reduction at all.
// P bounced via per-wave LDS (C-layout -> A-layout), P@V = 4 WMMAs.
// ---------------------------------------------------------------------------
__global__ __launch_bounds__(256, 2)
void attn_kernel(const bf16* __restrict__ Q,   // [B,h,N,64], pre-scaled
                 const bf16* __restrict__ Ks,  // [B,h,N,64]
                 const bf16* __restrict__ Vt,  // [B,h,64,N], row48 = ones
                 const float* __restrict__ tbl,// [63*63, 8]
                 bf16* __restrict__ Out) {     // [B,N,384] bf16
    __shared__ bf16 plds[8][16 * LPITCH];      // private 16x32 P tile per wave

    const int tid  = threadIdx.x;
    const int lane = tid & 31;
    const int wv   = tid >> 5;
    const int bh   = blockIdx.y;
    const int b    = bh >> 3, h = bh & 7;
    const int q0   = blockIdx.x * 128 + wv * 16;
    const int halfId = lane >> 4, l16 = lane & 15, kb = halfId * 8;

    const bf16* Qb = Q  + (long)bh * NTOK * QKDIM;
    const bf16* Kb = Ks + (long)bh * NTOK * QKDIM;
    const bf16* Vb = Vt + (long)bh * VROWS * NTOK;

    v16bf qa[2];
    {
        const bf16* qp = Qb + (long)(q0 + l16) * QKDIM;
        qa[0] = load_frag(qp + kb,      qp + kb + 16);
        qa[1] = load_frag(qp + 32 + kb, qp + 32 + kb + 16);
    }

    v8f o0 = {}, o1 = {}, o2 = {}, o3 = {};   // o3 col0 = softmax denominator
    float rmax[8];
#pragma unroll
    for (int r = 0; r < 8; ++r) rmax[r] = -3.0e38f;

    for (int j0 = 0; j0 < NTOK; j0 += 32) {
        // speculative prefetch of next key/value block (global_prefetch_b8;
        // harmlessly dropped past the end of the buffers)
        __builtin_prefetch(Kb + (long)(j0 + 32 + l16) * QKDIM, 0, 0);
        __builtin_prefetch(Vb + (long)l16 * NTOK + j0 + 32, 0, 0);

        v8f s[2];
#pragma unroll
        for (int jt = 0; jt < 2; ++jt) {
            const int col = j0 + jt * 16 + l16;
            const bf16* kp = Kb + (long)col * QKDIM;
            v16bf kf0 = load_frag(kp + kb,      kp + kb + 16);
            v16bf kf1 = load_frag(kp + 32 + kb, kp + 32 + kb + 16);
            v8f a = {};
            a = wmma_bf16(qa[0], kf0, a);
            a = wmma_bf16(qa[1], kf1, a);
            const int jh = col >> 5, jw = col & 31;
#pragma unroll
            for (int r = 0; r < 8; ++r) {
                const int qi = q0 + 8 * halfId + r;
                const int ih = qi >> 5, iw = qi & 31;
                a[r] += tbl[(((ih - jh + 31) * TBL) + (iw - jw + 31)) * NHEADS + h];
            }
            s[jt] = a;
        }

        float corr[8];
#pragma unroll
        for (int r = 0; r < 8; ++r) {
            const float m = red16_max(fmaxf(s[0][r], s[1][r]));
            const float nm = fmaxf(rmax[r], m);
            corr[r] = __expf(rmax[r] - nm);
            rmax[r] = nm;
            s[0][r] = __expf(s[0][r] - nm);
            s[1][r] = __expf(s[1][r] - nm);
        }
#pragma unroll
        for (int r = 0; r < 8; ++r) {
            o0[r] *= corr[r]; o1[r] *= corr[r];
            o2[r] *= corr[r]; o3[r] *= corr[r];
        }

        // P: C-layout regs -> LDS -> A-layout fragment (same-wave DS in-order)
        bf16* P = plds[wv];
#pragma unroll
        for (int r = 0; r < 8; ++r) {
            const int row = r + 8 * halfId;
            P[row * LPITCH + l16]      = (bf16)s[0][r];
            P[row * LPITCH + 16 + l16] = (bf16)s[1][r];
        }
        __builtin_amdgcn_wave_barrier();
        v16bf pf = load_frag(&P[l16 * LPITCH + kb], &P[l16 * LPITCH + kb + 16]);
        __builtin_amdgcn_wave_barrier();

#pragma unroll
        for (int t = 0; t < 4; ++t) {
            const bf16* vp = Vb + (long)(t * 16 + l16) * NTOK + j0;
            v16bf vf = load_frag(vp + kb, vp + kb + 16);
            if (t == 0)      o0 = wmma_bf16(pf, vf, o0);
            else if (t == 1) o1 = wmma_bf16(pf, vf, o1);
            else if (t == 2) o2 = wmma_bf16(pf, vf, o2);
            else             o3 = wmma_bf16(pf, vf, o3);
        }
    }

    bf16* ob = Out + (long)b * NTOK * CCH;
#pragma unroll
    for (int r = 0; r < 8; ++r) {
        // softmax denominator sits in o3 column 0 of each 16-lane group
        const float denom = __shfl(o3[r], halfId << 4, 32);
        const float inv = 1.0f / denom;
        const int row = q0 + 8 * halfId + r;
        const long base = (long)row * CCH + h * HDIM;
        ob[base + l16]      = (bf16)(o0[r] * inv);
        ob[base + 16 + l16] = (bf16)(o1[r] * inv);
        ob[base + 32 + l16] = (bf16)(o2[r] * inv);
    }
}

// ---------------------------------------------------------------------------
// Host-side orchestration
// ---------------------------------------------------------------------------
static inline size_t alignup(size_t v) { return (v + 255) & ~(size_t)255; }

extern "C" void kernel_launch(void* const* d_in, const int* in_sizes, int n_in,
                              void* d_out, int out_size, void* d_ws, size_t ws_size,
                              hipStream_t stream) {
    const float* x      = (const float*)d_in[0];
    const float* qkv_w  = (const float*)d_in[1];
    const float* qkv_b  = (const float*)d_in[2];
    const float* proj_w = (const float*)d_in[3];
    const float* proj_b = (const float*)d_in[4];
    const float* mlp_w1 = (const float*)d_in[5];
    const float* mlp_b1 = (const float*)d_in[6];
    const float* mlp_w2 = (const float*)d_in[7];
    const float* mlp_b2 = (const float*)d_in[8];

    char* ws = (char*)d_ws;
    size_t off = 0;
    bf16*  tok    = (bf16*)(ws + off); off = alignup(off + (size_t)BATCH * NTOK * CCH * 2);
    bf16*  qkvw_h = (bf16*)(ws + off); off = alignup(off + (size_t)CCH * 3 * CCH * 2);
    bf16*  projw_h= (bf16*)(ws + off); off = alignup(off + (size_t)CCH * CCH * 2);
    float* qkv    = (float*)(ws + off); off = alignup(off + (size_t)BATCH * NTOK * 3 * CCH * 4);
    bf16*  Qs     = (bf16*)(ws + off); off = alignup(off + (size_t)BATCH * NHEADS * NTOK * QKDIM * 2);
    bf16*  Kst    = (bf16*)(ws + off); off = alignup(off + (size_t)BATCH * NHEADS * NTOK * QKDIM * 2);
    bf16*  Vt     = (bf16*)(ws + off); off = alignup(off + (size_t)BATCH * NHEADS * VROWS * NTOK * 2);
    bf16*  attn   = (bf16*)(ws + off); off = alignup(off + (size_t)BATCH * NTOK * CCH * 2);
    float* tbl    = (float*)(ws + off); off = alignup(off + (size_t)TBL * TBL * NHEADS * 4);
    (void)ws_size; (void)in_sizes; (void)n_in; (void)out_size;

    // prep
    {
        int n = BATCH * NTOK * CCH;
        pack_tokens<<<(n + 255) / 256, 256, 0, stream>>>(x, tok);
    }
    {
        int n = CCH * 3 * CCH;
        convert_f32_bf16<<<(n + 255) / 256, 256, 0, stream>>>(qkv_w, qkvw_h, n);
        int m = CCH * CCH;
        convert_f32_bf16<<<(m + 255) / 256, 256, 0, stream>>>(proj_w, projw_h, m);
    }
    bias_table_kernel<<<(TBL * TBL + 255) / 256, 256, 0, stream>>>(
        mlp_w1, mlp_b1, mlp_w2, mlp_b2, tbl);

    // QKV GEMM: per batch [1024,384] @ [384,1152] + qkv_b
    {
        dim3 grid((NTOK / GBM) * (3 * CCH / GBN), BATCH);
        gemm_bf16_kernel<3 * CCH, CCH, false><<<grid, 256, 0, stream>>>(
            tok, qkvw_h, qkv_b, qkv,
            (long)NTOK * CCH, (long)NTOK * 3 * CCH, 0);
    }

    // split into scaled-Q / padded-K / transposed-V (with ones row)
    {
        int n = BATCH * NHEADS * NTOK * QKDIM;   // 2^22
        split_qkv_kernel<<<(n + 255) / 256, 256, 0, stream>>>(qkv, Qs, Kst, Vt);
    }

    // attention
    {
        dim3 grid(NTOK / 128, BATCH * NHEADS);
        attn_kernel<<<grid, 256, 0, stream>>>(Qs, Kst, Vt, tbl, attn);
    }

    // projection + transposed store straight into d_out [B,C,H,W]
    {
        dim3 grid((NTOK / GBM) * (CCH / GBN), BATCH);
        gemm_bf16_kernel<CCH, CCH, true><<<grid, 256, 0, stream>>>(
            attn, projw_h, proj_b, (float*)d_out,
            (long)NTOK * CCH, (long)CCH * NTOK, NTOK);
    }
}
